// SelfAttention_43078521979111
// MI455X (gfx1250) — compile-verified, matrix operand order checked by reading, not run on hardware
//
#include <hip/hip_runtime.h>

// ---------------------------------------------------------------------------
// CDNA5 (gfx1250) self-attention pipeline, bf16 WMMA with f32 accumulation.
// D = A(16x32 bf16) * B(32x16 bf16) + C(16x16 f32) via
// __builtin_amdgcn_wmma_f32_16x16x32_bf16  (wave32), plus async global->LDS
// staging (ASYNCcnt) for the attention V tiles.
// ---------------------------------------------------------------------------

typedef __attribute__((ext_vector_type(16))) __bf16 v16bf;
typedef __attribute__((ext_vector_type(8)))  __bf16 v8bf;
typedef __attribute__((ext_vector_type(8)))  float  v8f;

#define HW_   4096
#define C_    1280
#define CQK_  160
#define DOUT_ 1024
#define B_    4
#define MTOT_ (B_ * HW_)

static __device__ inline __bf16 f2bf(float f) {
  unsigned u = __builtin_bit_cast(unsigned, f);
  unsigned r = u + 0x7FFFu + ((u >> 16) & 1u);        // round-to-nearest-even
  unsigned short h = (unsigned short)(r >> 16);
  return __builtin_bit_cast(__bf16, h);
}

static __device__ inline v8f wmma_bf16(v16bf a, v16bf b, v8f c) {
  return __builtin_amdgcn_wmma_f32_16x16x32_bf16(
      /*neg_a=*/false, a, /*neg_b=*/false, b,
      /*c_mod=*/(short)0, c, /*reuse_a=*/false, /*reuse_b=*/false);
}

static __device__ inline v8f zero8f() {
  v8f z = {0.f, 0.f, 0.f, 0.f, 0.f, 0.f, 0.f, 0.f};
  return z;
}

// Load a 16x32 bf16 fragment (A-operand layout; also yields B = G^T when fed
// rows of an (N x K) matrix G).  `tile` points at element [row 0, k 0],
// ld = row stride in elements.  Per ISA 7.12.2: lane holds row = lane%16;
// lanes 0-15 carry K {0..7, 16..23}, lanes 16-31 carry K {8..15, 24..31}.
static __device__ inline v16bf ldfrag(const __bf16* tile, int ld) {
  int lane = threadIdx.x & 31;
  int r  = lane & 15;
  int kh = (lane >> 4) << 3;                 // 0 or 8
  const v8bf* p0 = (const v8bf*)(tile + (size_t)r * ld + kh);
  const v8bf* p1 = (const v8bf*)(tile + (size_t)r * ld + kh + 16);
  v8bf a = *p0;
  v8bf b = *p1;
  return __builtin_shufflevector(a, b, 0,1,2,3,4,5,6,7,8,9,10,11,12,13,14,15);
}

static __device__ inline v16bf ldfrag_guard(const __bf16* tile, int ld, int rows_ok) {
  int lane = threadIdx.x & 31;
  int r  = lane & 15;
  int kh = (lane >> 4) << 3;
  v16bf f;
  if (r < rows_ok) {
    const v8bf* p0 = (const v8bf*)(tile + (size_t)r * ld + kh);
    const v8bf* p1 = (const v8bf*)(tile + (size_t)r * ld + kh + 16);
    v8bf a = *p0;
    v8bf b = *p1;
    f = __builtin_shufflevector(a, b, 0,1,2,3,4,5,6,7,8,9,10,11,12,13,14,15);
  } else {
#pragma unroll
    for (int i = 0; i < 16; ++i) f[i] = (__bf16)0.0f;
  }
  return f;
}

// ---------------------------------------------------------------------------
// fp32 -> bf16 conversion
// ---------------------------------------------------------------------------
__global__ void cvt_bf16_kernel(const float* __restrict__ s, __bf16* __restrict__ d, long n) {
  long i = (long)blockIdx.x * blockDim.x + threadIdx.x;
  long stride = (long)gridDim.x * blockDim.x;
  for (; i < n; i += stride) d[i] = f2bf(s[i]);
}

// ---------------------------------------------------------------------------
// Generic WMMA GEMM:  out(M x N) = A(M x K, bf16 rm) * G(N x K, bf16 rm)^T + bias
// mode 0: bf16 row-major out      mode 1: bf16 batched-transposed out
// mode 2: fp32 row-major out      (mode 1: out[b][n][m%4096], b = m/4096)
// One wave computes a 16 x 64 output tile (4 WMMA n-subtiles).  Full strips
// take a guard-free fast path (no EXEC juggling in the K loop).
// ---------------------------------------------------------------------------
__global__ __launch_bounds__(256) void gemm_wmma_kernel(
    const __bf16* __restrict__ A, const __bf16* __restrict__ G,
    const float* __restrict__ bias, void* __restrict__ dst,
    int M, int N, int K, int mode) {
  int wid  = (blockIdx.x * blockDim.x + threadIdx.x) >> 5;
  int lane = threadIdx.x & 31;
  int nstrips = (N + 63) >> 6;
  int totalW  = (M >> 4) * nstrips;
  if (wid >= totalW) return;
  int mt = wid / nstrips;
  int ns = wid % nstrips;

  const __bf16* Arow = A + (size_t)mt * 16 * K;
  const __bf16* Gst  = G + (size_t)ns * 64 * K;
  int rows_left = N - ns * 64;

  v8f acc[4];
#pragma unroll
  for (int s = 0; s < 4; ++s) acc[s] = zero8f();

  if (rows_left >= 64) {
    // Fast path: all four 16-row B panels in-bounds.
    for (int kc = 0; kc < K; kc += 32) {
      v16bf af = ldfrag(Arow + kc, K);
#pragma unroll
      for (int s = 0; s < 4; ++s) {
        v16bf bf = ldfrag(Gst + (size_t)s * 16 * K + kc, K);
        acc[s] = wmma_bf16(af, bf, acc[s]);
      }
    }
  } else {
    for (int kc = 0; kc < K; kc += 32) {
      v16bf af = ldfrag(Arow + kc, K);
#pragma unroll
      for (int s = 0; s < 4; ++s) {
        v16bf bf = ldfrag_guard(Gst + (size_t)s * 16 * K + kc, K, rows_left - s * 16);
        acc[s] = wmma_bf16(af, bf, acc[s]);
      }
    }
  }

  int hi = lane >> 4, col = lane & 15;
#pragma unroll
  for (int s = 0; s < 4; ++s) {
    int gn = ns * 64 + s * 16 + col;
    if (gn >= N) continue;
    float bval = bias ? bias[gn] : 0.0f;
#pragma unroll
    for (int v = 0; v < 8; ++v) {
      int gm = mt * 16 + v + 8 * hi;
      float val = acc[s][v] + bval;
      if (mode == 0) {
        ((__bf16*)dst)[(size_t)gm * N + gn] = f2bf(val);
      } else if (mode == 1) {
        ((__bf16*)dst)[((size_t)(gm >> 12) * N + gn) * 4096 + (gm & 4095)] = f2bf(val);
      } else {
        ((float*)dst)[(size_t)gm * N + gn] = val;
      }
    }
  }
}

// ---------------------------------------------------------------------------
// Flash-attention kernel.  Block = 256 threads (8 waves), 32 query rows.
// Per j-block of 32:
//   * all waves issue async global->LDS copies of the (1280 x 32) V slice
//     (global_load_async_to_lds_b128, ASYNCcnt) -- overlaps next two phases
//   * waves 0-3: S = Q K^T tiles (WMMA, CQK=160 contraction) + mask -> LDS
//   * wave 0:    online softmax row pass, P(bf16) + rescale factors -> LDS
//   * s_wait_asynccnt 0 + barrier, then all waves rescale O and accumulate
//     O += P * V^T (20 WMMA n-subtiles/wave, B-frags via ds_load_b128)
// Epilogue: out = gamma * (O / l) + x, stored bf16 for the final projection.
// ---------------------------------------------------------------------------
__global__ __launch_bounds__(256) void flash_attn_kernel(
    const __bf16* __restrict__ Q,     // (B*HW, CQK)
    const __bf16* __restrict__ Km,    // (B*HW, CQK)
    const __bf16* __restrict__ Vc,    // (B, C, HW)
    const int*    __restrict__ mask,  // (B, HW)
    const float*  __restrict__ x,     // (B, HW, C)
    const float*  __restrict__ gamma, // (1,)
    __bf16*       __restrict__ attn)  // (B*HW, C) bf16
{
  // V tile: 1280 rows x 32 cols, rows padded to 40 elements (80 B) so the
  // 16-lane b128 fragment reads tile the 64 LDS banks with no conflicts.
  __shared__ __align__(16) __bf16 VT[C_ * 40];        // 100 KB
  __shared__ float  Sh[32 * 33];                      // score staging (padded)
  __shared__ __align__(16) __bf16 Pl[32 * 40];        // probs, bf16 (padded)
  __shared__ float  RS[32];                           // per-row rescale factor
  __shared__ float  LL[32];                           // final softmax denoms

  int wid  = threadIdx.x >> 5;
  int lane = threadIdx.x & 31;
  int b     = blockIdx.x >> 7;        // 128 blocks per batch (4096/32)
  int qrow0 = (blockIdx.x & 127) * 32;

  const __bf16* Qb  = Q  + (size_t)b * HW_ * CQK_;
  const __bf16* Kb  = Km + (size_t)b * HW_ * CQK_;
  const __bf16* Vb  = Vc + (size_t)b * C_ * HW_;
  const int*    mkb = mask + (size_t)b * HW_;
  const float*  xb  = x    + (size_t)b * HW_ * C_;
  __bf16*       ab  = attn + (size_t)(b * HW_ + qrow0) * C_;

  unsigned vt_base = (unsigned)(uintptr_t)(&VT[0]);   // LDS byte offset

  // O accumulators: this wave owns rows [omt*16, omt*16+16) x cols [c0, c0+320)
  int omt = wid >> 2;
  int c0  = (wid & 3) * 320;
  v8f oacc[20];
#pragma unroll
  for (int t = 0; t < 20; ++t) oacc[t] = zero8f();

  // Waves 0-3 preload their Q fragments (16 rows x 160 = 5 k-fragments).
  v16bf qf[5];
  if (wid < 4) {
    int mt = wid >> 1;
#pragma unroll
    for (int kc = 0; kc < 5; ++kc)
      qf[kc] = ldfrag(Qb + (size_t)(qrow0 + mt * 16) * CQK_ + kc * 32, CQK_);
  }

  float m_run = -3.0e38f;   // per-row running max (wave 0, lane = row)
  float l_run = 0.0f;       // per-row running denom

  for (int jb = 0; jb < HW_; jb += 32) {
    // ---- async-copy the V j-slice into LDS (overlaps S + softmax) ----
    // 1280 rows x 4 x b128 chunks = 5120 transfers, 20 per thread.
    {
      const __bf16* vsrc = Vb + jb;
      int tid = threadIdx.x;
#pragma unroll
      for (int i = 0; i < 20; ++i) {
        int cidx = tid + i * 256;             // 0..5119
        int row  = cidx >> 2;
        int ch   = cidx & 3;
        unsigned ldsa = vt_base + (unsigned)(row * 80 + ch * 16);
        unsigned long long ga =
            (unsigned long long)(uintptr_t)(vsrc + (size_t)row * HW_ + ch * 8);
        asm volatile("global_load_async_to_lds_b128 %0, %1, off"
                     :: "v"(ldsa), "v"(ga) : "memory");
      }
    }

    // ---- S = Q K^T for a 32x32 tile, mask folded in at the LDS store ----
    if (wid < 4) {
      int mt = wid >> 1, nt = wid & 1;
      v8f s = zero8f();
#pragma unroll
      for (int kc = 0; kc < 5; ++kc) {
        v16bf kf = ldfrag(Kb + (size_t)(jb + nt * 16) * CQK_ + kc * 32, CQK_);
        s = wmma_bf16(qf[kc], kf, s);
      }
      int col = lane & 15, hi = lane >> 4;
      float madd = (mkb[jb + nt * 16 + col] != 0) ? 0.0f : -1000000.0f;
#pragma unroll
      for (int v = 0; v < 8; ++v)
        Sh[(mt * 16 + v + 8 * hi) * 33 + nt * 16 + col] = s[v] + madd;
    }
    __syncthreads();

    // ---- online softmax (wave 0, one lane per query row) ----
    if (wid == 0) {
      const float* row = &Sh[lane * 33];
      float mx = m_run;
#pragma unroll 8
      for (int c = 0; c < 32; ++c) mx = fmaxf(mx, row[c]);
      float rs  = __expf(m_run - mx);
      float sum = 0.0f;
#pragma unroll 8
      for (int c = 0; c < 32; ++c) {
        float e = __expf(row[c] - mx);
        sum += e;
        Pl[lane * 40 + c] = f2bf(e);
      }
      l_run = l_run * rs + sum;
      m_run = mx;
      RS[lane] = rs;
    }

    // Drain this wave's async V copies, then make them (and P) visible.
    asm volatile("s_wait_asynccnt 0x0" ::: "memory");
    __syncthreads();

    // ---- O = O * rs + P * V^T  (all 8 waves, V from LDS) ----
    {
      int hi = lane >> 4;
      float rv[8];
#pragma unroll
      for (int v = 0; v < 8; ++v) rv[v] = RS[omt * 16 + 8 * hi + v];
#pragma unroll
      for (int t = 0; t < 20; ++t)
#pragma unroll
        for (int v = 0; v < 8; ++v) oacc[t][v] *= rv[v];

      v16bf pf = ldfrag(&Pl[(omt * 16) * 40], 40);   // A-frag from LDS
#pragma unroll
      for (int t = 0; t < 20; ++t) {
        v16bf vf = ldfrag(&VT[(c0 + t * 16) * 40], 40);  // B-frag from LDS
        oacc[t] = wmma_bf16(pf, vf, oacc[t]);
      }
    }
    __syncthreads();    // protect VT/Sh/Pl before next iteration overwrites
  }

  if (wid == 0) LL[lane] = l_run;
  __syncthreads();

  // ---- epilogue: normalize, gamma-residual, store bf16 ----
  {
    int hi = lane >> 4, col = lane & 15;
    float g = gamma[0];
    float ll[8];
#pragma unroll
    for (int v = 0; v < 8; ++v) ll[v] = LL[omt * 16 + 8 * hi + v];
#pragma unroll
    for (int t = 0; t < 20; ++t) {
      int gc = c0 + t * 16 + col;
#pragma unroll
      for (int v = 0; v < 8; ++v) {
        int r32 = omt * 16 + 8 * hi + v;
        float val = oacc[t][v] / ll[v];
        float res = g * val + xb[(size_t)(qrow0 + r32) * C_ + gc];
        ab[(size_t)r32 * C_ + gc] = f2bf(res);
      }
    }
  }
}

// ---------------------------------------------------------------------------
// Host-side orchestration
// ---------------------------------------------------------------------------
extern "C" void kernel_launch(void* const* d_in, const int* in_sizes, int n_in,
                              void* d_out, int out_size, void* d_ws, size_t ws_size,
                              hipStream_t stream) {
  (void)in_sizes; (void)n_in; (void)out_size; (void)ws_size;
  const float* x     = (const float*)d_in[0];
  const int*   mask  = (const int*)  d_in[1];
  const float* Wq    = (const float*)d_in[2];
  const float* bq    = (const float*)d_in[3];
  const float* Wk    = (const float*)d_in[4];
  const float* bk    = (const float*)d_in[5];
  const float* Wv    = (const float*)d_in[6];
  const float* bv    = (const float*)d_in[7];
  const float* gamma = (const float*)d_in[8];
  const float* Wp    = (const float*)d_in[9];
  const float* bp    = (const float*)d_in[10];
  float* out = (float*)d_out;

  // Workspace carve-out (bf16 tensors), 256B aligned.
  char* ws = (char*)d_ws;
  size_t off = 0;
  auto carve = [&](size_t bytes) -> void* {
    off = (off + 255) & ~(size_t)255;
    void* p = ws + off;
    off += bytes;
    return p;
  };
  __bf16* xb   = (__bf16*)carve((size_t)MTOT_ * C_ * 2);
  __bf16* wqb  = (__bf16*)carve((size_t)CQK_ * C_ * 2);
  __bf16* wkb  = (__bf16*)carve((size_t)CQK_ * C_ * 2);
  __bf16* wvb  = (__bf16*)carve((size_t)C_ * C_ * 2);
  __bf16* wpb  = (__bf16*)carve((size_t)DOUT_ * C_ * 2);
  __bf16* Qb   = (__bf16*)carve((size_t)MTOT_ * CQK_ * 2);
  __bf16* Kb   = (__bf16*)carve((size_t)MTOT_ * CQK_ * 2);
  __bf16* Vc   = (__bf16*)carve((size_t)B_ * C_ * HW_ * 2);   // (B, C, HW)
  __bf16* attn = (__bf16*)carve((size_t)MTOT_ * C_ * 2);

  auto cvt = [&](const float* s, __bf16* d, long n) {
    int blocks = (int)((n + 255) / 256);
    if (blocks > 4096) blocks = 4096;
    cvt_bf16_kernel<<<blocks, 256, 0, stream>>>(s, d, n);
  };
  cvt(x,  xb,  (long)MTOT_ * C_);
  cvt(Wq, wqb, (long)CQK_ * C_);
  cvt(Wk, wkb, (long)CQK_ * C_);
  cvt(Wv, wvb, (long)C_ * C_);
  cvt(Wp, wpb, (long)DOUT_ * C_);

  auto gemm = [&](const __bf16* A, const __bf16* G, const float* bias,
                  void* dst, int M, int N, int K, int mode) {
    int nstrips = (N + 63) >> 6;
    long totalW = (long)(M >> 4) * nstrips;
    int blocks  = (int)((totalW * 32 + 255) / 256);
    gemm_wmma_kernel<<<blocks, 256, 0, stream>>>(A, G, bias, dst, M, N, K, mode);
  };

  // Projections: Q, K (bf16 rm), V (bf16 transposed per batch -> (B, C, HW)).
  gemm(xb, wqb, bq, Qb, MTOT_, CQK_, C_, 0);
  gemm(xb, wkb, bk, Kb, MTOT_, CQK_, C_, 0);
  gemm(xb, wvb, bv, Vc, MTOT_, C_,   C_, 1);

  // Flash attention: 128 blocks per batch (32 query rows each).
  flash_attn_kernel<<<B_ * (HW_ / 32), 256, 0, stream>>>(
      Qb, Kb, Vc, mask, x, gamma, attn);

  // Final projection to fp32 output.
  gemm(attn, wpb, bp, out, MTOT_, DOUT_, C_, 2);
}
